// PointTransformerLayer_55198919688302
// MI455X (gfx1250) — matrix operand order repaired and checked by reading, not compiled, not run on hardware
//
#include <hip/hip_runtime.h>
#include <hip/hip_bf16.h>
#include <math.h>

#define MPTS 8192
#define NPTS 8192
#define BATCH 4
#define KNN 16
#define FDIM 64
#define HID 128
#define AST 136   // padded LDS stride (f16 elems) for 128-deep activation tiles
#define XST 72    // padded LDS stride (f16 elems) for 64-deep qkv input tiles

typedef __attribute__((ext_vector_type(16))) _Float16 v16h;
typedef __attribute__((ext_vector_type(8)))  _Float16 v8h;
typedef __attribute__((ext_vector_type(8)))  float    v8f;

// ---- WMMA fragment helpers (gfx1250 wave32, V_WMMA_F32_16X16X32_F16) ----

// A-matrix 16x32 from f16 row-major weights: two 16B vector loads per lane.
__device__ __forceinline__ v16h load_afrag_h(const _Float16* Wh, int ldw, int row0, int k0, int lane) {
  const _Float16* p = Wh + (size_t)(row0 + (lane & 15)) * ldw + k0 + ((lane >> 4) << 3);
  v8h a0 = *(const v8h*)p;
  v8h a1 = *(const v8h*)(p + 16);
  return __builtin_shufflevector(a0, a1, 0,1,2,3,4,5,6,7,8,9,10,11,12,13,14,15);
}

// B-matrix 32x16 from transposed LDS activation tile[n*stride + kdim]:
// lane<16 -> N=lane, K=k0+0..7,16..23 ; lane>=16 -> N=lane-16, K=k0+8..15,24..31. Two 16B LDS loads.
__device__ __forceinline__ v16h load_bfrag_T(const _Float16* tileT, int stride, int k0, int lane) {
  const _Float16* p = tileT + (lane & 15) * stride + k0 + ((lane >> 4) << 3);
  v8h b0 = *(const v8h*)p;
  v8h b1 = *(const v8h*)(p + 16);
  return __builtin_shufflevector(b0, b1, 0,1,2,3,4,5,6,7,8,9,10,11,12,13,14,15);
}

__device__ __forceinline__ float wred16(float v) {
#pragma unroll
  for (int d = 1; d < 16; d <<= 1) v += __shfl_xor(v, d, 16);
  return v;
}
__device__ __forceinline__ float wmax16(float v) {
#pragma unroll
  for (int d = 1; d < 16; d <<= 1) v = fmaxf(v, __shfl_xor(v, d, 16));
  return v;
}
__device__ __forceinline__ float wred32(float v) {
#pragma unroll
  for (int d = 1; d < 32; d <<= 1) v += __shfl_xor(v, d, 32);
  return v;
}

// ---- Kernel 0: convert all GEMM weights fp32 -> f16 once ----
__global__ __launch_bounds__(256) void k_cvtw(
    const float* wq, const float* wk, const float* wv,
    const float* d2, const float* g1, const float* g2, _Float16* wh)
{
  int i = blockIdx.x * 256 + threadIdx.x;
  const int SQ = HID * FDIM;  // 8192
  const int SB = HID * HID;   // 16384
  if (i < SQ)                   wh[i] = (_Float16)wq[i];
  else if (i < 2 * SQ)          wh[i] = (_Float16)wk[i - SQ];
  else if (i < 3 * SQ)          wh[i] = (_Float16)wv[i - 2 * SQ];
  else if (i < 3 * SQ + SB)     wh[i] = (_Float16)d2[i - 3 * SQ];
  else if (i < 3 * SQ + 2 * SB) wh[i] = (_Float16)g1[i - 3 * SQ - SB];
  else if (i < 3 * SQ + 3 * SB) wh[i] = (_Float16)g2[i - 3 * SQ - 2 * SB];
}

// ---- Kernel 1: q/k/v projections. Y[pt][c] f16 = W(128x64) X(64 x pts) + b ----
// Cooperative LDS stage of the 64x16 X tile (shared by all 8 row-tile waves).
__global__ __launch_bounds__(256) void k_qkv(
    const _Float16* Wh0, const float* B0, const float* X0, _Float16* Y0,
    const _Float16* Wh1, const float* B1, const float* X1, _Float16* Y1,
    const _Float16* Wh2, const float* B2, const float* X2, _Float16* Y2)
{
  __shared__ _Float16 xT[16 * XST];
  int z = blockIdx.z;
  const _Float16* W = (z == 0) ? Wh0 : ((z == 1) ? Wh1 : Wh2);
  const float* Bv   = (z == 0) ? B0 : ((z == 1) ? B1 : B2);
  const float* X    = (z == 0) ? X0 : ((z == 1) ? X1 : X2);
  _Float16* Y       = (z == 0) ? Y0 : ((z == 1) ? Y1 : Y2);
  int b = blockIdx.y;
  int col0 = blockIdx.x * 16;
  int tid = threadIdx.x;
  int lane = tid & 31;
  int wave = tid >> 5;
  const float* Xb = X + (size_t)b * FDIM * NPTS;
  // coalesced fp32 load -> f16 transposed tile
  for (int e = tid; e < FDIM * 16; e += 256) {
    int c = e >> 4, n = e & 15;
    xT[n * XST + c] = (_Float16)Xb[(size_t)c * NPTS + col0 + n];
  }
  __syncthreads();

  int row0 = wave * 16;
  int half8 = (lane >> 4) << 3;
  int rb = row0 + half8;
  v8f acc;
#pragma unroll
  for (int j = 0; j < 8; ++j) acc[j] = Bv[rb + j];
#pragma unroll
  for (int s = 0; s < 2; ++s) {
    v16h a  = load_afrag_h(W, FDIM, row0, s * 32, lane);
    v16h bf = load_bfrag_T(xT, XST, s * 32, lane);
    acc = __builtin_amdgcn_wmma_f32_16x16x32_f16(false, a, false, bf, (short)0, acc, false, false);
  }
  int n = lane & 15;
  v8h o;
#pragma unroll
  for (int j = 0; j < 8; ++j) o[j] = (_Float16)acc[j];
  *(v8h*)(Y + ((size_t)b * NPTS + col0 + n) * HID + rb) = o;
}

// ---- Kernel 2: per-batch first/second moments of rel (analytic dgn stats) ----
__global__ __launch_bounds__(256) void k_relstats(
    const float* qx, const float* kx, const int* idx, float* out)
{
  int b = blockIdx.y;
  float s[9];
#pragma unroll
  for (int i = 0; i < 9; ++i) s[i] = 0.f;
  const float* qxb = qx + (size_t)b * 3 * MPTS;
  const float* kxb = kx + (size_t)b * 3 * NPTS;
  const int* idxb = idx + (size_t)b * MPTS * KNN;
  for (int e = blockIdx.x * blockDim.x + threadIdx.x; e < MPTS * KNN;
       e += gridDim.x * blockDim.x) {
    int m = e >> 4, k = e & 15;
    int id = idxb[m * KNN + k];
    float r0 = qxb[m]          - kxb[id];
    float r1 = qxb[MPTS + m]   - kxb[NPTS + id];
    float r2 = qxb[2*MPTS + m] - kxb[2*NPTS + id];
    s[0]+=r0; s[1]+=r1; s[2]+=r2;
    s[3]+=r0*r0; s[4]+=r0*r1; s[5]+=r0*r2; s[6]+=r1*r1; s[7]+=r1*r2; s[8]+=r2*r2;
  }
  int lane = threadIdx.x & 31;
#pragma unroll
  for (int i = 0; i < 9; ++i) {
    float v = wred32(s[i]);
    if (lane == 0) atomicAdd(&out[b * 9 + i], v);
  }
}

// ---- Kernel 3: dgn group stats from rel moments ----
__global__ void k_dgn(const float* rs, const float* d1w, const float* d1b, float* st)
{
  int t = threadIdx.x;
  if (t >= BATCH * 8) return;
  int b = t >> 3, g = t & 7;
  float inv = 1.f / (float)(MPTS * KNN);
  float Ex0 = rs[b*9+0]*inv, Ex1 = rs[b*9+1]*inv, Ex2 = rs[b*9+2]*inv;
  float E00 = rs[b*9+3]*inv, E01 = rs[b*9+4]*inv, E02 = rs[b*9+5]*inv;
  float E11 = rs[b*9+6]*inv, E12 = rs[b*9+7]*inv, E22 = rs[b*9+8]*inv;
  float sum = 0.f, sum2 = 0.f;
  for (int c = g*16; c < g*16+16; ++c) {
    float w0 = d1w[c*3], w1 = d1w[c*3+1], w2 = d1w[c*3+2], bb = d1b[c];
    float wx = w0*Ex0 + w1*Ex1 + w2*Ex2;
    float ey = wx + bb;
    float q  = w0*w0*E00 + w1*w1*E11 + w2*w2*E22
             + 2.f*(w0*w1*E01 + w0*w2*E02 + w1*w2*E12);
    float ey2 = q + 2.f*bb*wx + bb*bb;
    sum += ey; sum2 += ey2;
  }
  float mu  = sum * (1.f/16.f);
  float var = sum2 * (1.f/16.f) - mu*mu;
  st[t*2]   = mu;
  st[t*2+1] = rsqrtf(var + 1e-5f);
}

// ---- Kernel 4: STATS ONLY pass. pos_enc -> attn_in -> h = g1*attn_in; accumulate
// per-(b,group) sum/sumsq of h. Nothing stored to HBM (h is recomputed later). ----
__global__ __launch_bounds__(256) void k_stats(
    const float* qx, const float* kx, const int* idx,
    const float* d1w, const float* d1b, const float* dgnw, const float* dgnb,
    const _Float16* wh_d2, const float* d2b,
    const _Float16* wh_g1, const float* g1b,
    const _Float16* q_h, const _Float16* k_h,
    const float* dgn_st, float* ggn_sums)
{
  __shared__ int idx_s[8][KNN];
  __shared__ float rel_s[8][3][KNN];
  __shared__ _Float16 actT[8][KNN * AST];
  __shared__ float gsum[8], gsum2[8];

  int tid = threadIdx.x;
  int wave = tid >> 5, lane = tid & 31;
  int b = blockIdx.y;
  int m = blockIdx.x * 8 + wave;

  if (tid < 8) { gsum[tid] = 0.f; gsum2[tid] = 0.f; }

  const float* qxb = qx + (size_t)b*3*MPTS;
  const float* kxb = kx + (size_t)b*3*NPTS;
  if (lane < 16) {
    int id = idx[((size_t)b*MPTS + m)*KNN + lane];
    idx_s[wave][lane] = id;
    rel_s[wave][0][lane] = qxb[m]          - kxb[id];
    rel_s[wave][1][lane] = qxb[MPTS+m]     - kxb[NPTS+id];
    rel_s[wave][2][lane] = qxb[2*MPTS+m]   - kxb[2*NPTS+id];
  }
  __syncthreads();

  for (int e = lane; e < HID*KNN; e += 32) {
    int c = e & 127, k = e >> 7;
    float v = d1w[c*3]*rel_s[wave][0][k] + d1w[c*3+1]*rel_s[wave][1][k]
            + d1w[c*3+2]*rel_s[wave][2][k] + d1b[c];
    int g = c >> 4;
    float mu = dgn_st[(b*8+g)*2], rsg = dgn_st[(b*8+g)*2+1];
    v = (v - mu) * rsg * dgnw[c] + dgnb[c];
    actT[wave][k*AST + c] = (_Float16)fmaxf(v, 0.f);
  }
  __syncthreads();

  v16h bf[4];
#pragma unroll
  for (int s = 0; s < 4; ++s) bf[s] = load_bfrag_T(actT[wave], AST, s*32, lane);

  int half8 = (lane >> 4) << 3;
  int kcol = lane & 15;
  v8f pacc[8];
#pragma unroll
  for (int rt = 0; rt < 8; ++rt) {
    v8f acc;
#pragma unroll
    for (int j = 0; j < 8; ++j) acc[j] = d2b[rt*16 + half8 + j];
#pragma unroll
    for (int s = 0; s < 4; ++s) {
      v16h a = load_afrag_h(wh_d2, HID, rt*16, s*32, lane);
      acc = __builtin_amdgcn_wmma_f32_16x16x32_f16(false, a, false, bf[s], (short)0, acc, false, false);
    }
    pacc[rt] = acc;
  }
  __syncthreads();

  int id_k = idx_s[wave][kcol];
  const _Float16* qrow = q_h + ((size_t)b*MPTS + m)*HID;
  const _Float16* krow = k_h + ((size_t)b*NPTS + id_k)*HID;
#pragma unroll
  for (int rt = 0; rt < 8; ++rt) {
    int rb = rt*16 + half8;
    v8h q8 = *(const v8h*)(qrow + rb);
    v8h k8 = *(const v8h*)(krow + rb);
    v8h ai8;
#pragma unroll
    for (int j = 0; j < 8; ++j)
      ai8[j] = (_Float16)((float)q8[j] - (float)k8[j] + pacc[rt][j]);
    *(v8h*)(&actT[wave][kcol*AST + rb]) = ai8;
  }
  __syncthreads();

#pragma unroll
  for (int s = 0; s < 4; ++s) bf[s] = load_bfrag_T(actT[wave], AST, s*32, lane);
#pragma unroll
  for (int rt = 0; rt < 8; ++rt) {
    v8f acc;
#pragma unroll
    for (int j = 0; j < 8; ++j) acc[j] = g1b[rt*16 + half8 + j];
#pragma unroll
    for (int s = 0; s < 4; ++s) {
      v16h a = load_afrag_h(wh_g1, HID, rt*16, s*32, lane);
      acc = __builtin_amdgcn_wmma_f32_16x16x32_f16(false, a, false, bf[s], (short)0, acc, false, false);
    }
    float s1 = 0.f, s2 = 0.f;
#pragma unroll
    for (int j = 0; j < 8; ++j) { float h = acc[j]; s1 += h; s2 += h*h; }
    s1 = wred32(s1); s2 = wred32(s2);
    if (lane == 0) { atomicAdd(&gsum[rt], s1); atomicAdd(&gsum2[rt], s2); }
  }
  __syncthreads();
  if (tid < 8) {
    atomicAdd(&ggn_sums[(b*8+tid)*2],   gsum[tid]);
    atomicAdd(&ggn_sums[(b*8+tid)*2+1], gsum2[tid]);
  }
}

// ---- Kernel 5: finalize ggn stats ----
__global__ void k_ggnfin(const float* sums, float* st)
{
  int t = threadIdx.x;
  if (t >= BATCH*8) return;
  float cnt = 16.f * (float)MPTS * (float)KNN;
  float mu  = sums[t*2] / cnt;
  float var = sums[t*2+1] / cnt - mu*mu;
  st[t*2]   = mu;
  st[t*2+1] = rsqrtf(var + 1e-5f);
}

// ---- Kernel 6: full recompute pass. pos_enc (kept in regs) -> attn_in -> h (g1) ->
// GN(stats)+relu in regs -> g2 -> softmax over K -> sum attn*(value+pos) -> post conv + residual ----
__global__ __launch_bounds__(256) void k_final(
    const float* qx, const float* kx, const int* idx, const unsigned char* mask,
    const float* d1w, const float* d1b, const float* dgnw, const float* dgnb,
    const _Float16* wh_d2, const float* d2b,
    const _Float16* wh_g1, const float* g1b,
    const float* ggnw, const float* ggnb,
    const _Float16* wh_g2, const float* g2b,
    const float* postw, const float* postb,
    const float* q_feats,
    const _Float16* q_h, const _Float16* k_h, const _Float16* v_h,
    const float* dgn_st, const float* ggn_st,
    float* out)
{
  __shared__ int idx_s[8][KNN];
  __shared__ float rel_s[8][3][KNN];
  __shared__ _Float16 actT[8][KNN * AST];
  __shared__ float res_s[8][HID];

  int tid = threadIdx.x;
  int wave = tid >> 5, lane = tid & 31;
  int b = blockIdx.y;
  int m = blockIdx.x*8 + wave;

  const float* qxb = qx + (size_t)b*3*MPTS;
  const float* kxb = kx + (size_t)b*3*NPTS;
  if (lane < 16) {
    int id = idx[((size_t)b*MPTS + m)*KNN + lane];
    idx_s[wave][lane] = id;
    rel_s[wave][0][lane] = qxb[m]          - kxb[id];
    rel_s[wave][1][lane] = qxb[MPTS+m]     - kxb[NPTS+id];
    rel_s[wave][2][lane] = qxb[2*MPTS+m]   - kxb[2*NPTS+id];
  }
  __syncthreads();

  // act1 = relu(GN(d1*rel + b))
  for (int e = lane; e < HID*KNN; e += 32) {
    int c = e & 127, k = e >> 7;
    float v = d1w[c*3]*rel_s[wave][0][k] + d1w[c*3+1]*rel_s[wave][1][k]
            + d1w[c*3+2]*rel_s[wave][2][k] + d1b[c];
    int g = c >> 4;
    float mu = dgn_st[(b*8+g)*2], rsg = dgn_st[(b*8+g)*2+1];
    v = (v - mu) * rsg * dgnw[c] + dgnb[c];
    actT[wave][k*AST + c] = (_Float16)fmaxf(v, 0.f);
  }
  __syncthreads();

  v16h bf[4];
#pragma unroll
  for (int s = 0; s < 4; ++s) bf[s] = load_bfrag_T(actT[wave], AST, s*32, lane);

  int half8 = (lane >> 4) << 3;
  int kcol = lane & 15;
  // pos_enc = d2 * act1 + b : keep in registers for the whole kernel
  v8f pacc[8];
#pragma unroll
  for (int rt = 0; rt < 8; ++rt) {
    v8f acc;
#pragma unroll
    for (int j = 0; j < 8; ++j) acc[j] = d2b[rt*16 + half8 + j];
#pragma unroll
    for (int s = 0; s < 4; ++s) {
      v16h a = load_afrag_h(wh_d2, HID, rt*16, s*32, lane);
      acc = __builtin_amdgcn_wmma_f32_16x16x32_f16(false, a, false, bf[s], (short)0, acc, false, false);
    }
    pacc[rt] = acc;
  }
  __syncthreads();

  int id_k = idx_s[wave][kcol];
  const _Float16* qrow = q_h + ((size_t)b*MPTS + m)*HID;
  const _Float16* krow = k_h + ((size_t)b*NPTS + id_k)*HID;
#pragma unroll
  for (int rt = 0; rt < 8; ++rt) {
    int rb = rt*16 + half8;
    v8h q8 = *(const v8h*)(qrow + rb);
    v8h k8 = *(const v8h*)(krow + rb);
    v8h ai8;
#pragma unroll
    for (int j = 0; j < 8; ++j)
      ai8[j] = (_Float16)((float)q8[j] - (float)k8[j] + pacc[rt][j]);
    *(v8h*)(&actT[wave][kcol*AST + rb]) = ai8;
  }
  __syncthreads();

  // h = g1*attn_in + b, then GN (global stats) + relu applied in registers -> actT
#pragma unroll
  for (int s = 0; s < 4; ++s) bf[s] = load_bfrag_T(actT[wave], AST, s*32, lane);
#pragma unroll
  for (int rt = 0; rt < 8; ++rt) {
    v8f acc;
#pragma unroll
    for (int j = 0; j < 8; ++j) acc[j] = g1b[rt*16 + half8 + j];
#pragma unroll
    for (int s = 0; s < 4; ++s) {
      v16h a = load_afrag_h(wh_g1, HID, rt*16, s*32, lane);
      acc = __builtin_amdgcn_wmma_f32_16x16x32_f16(false, a, false, bf[s], (short)0, acc, false, false);
    }
    // group of rows rt*16..rt*16+15 is exactly group rt
    float mu = ggn_st[(b*8+rt)*2], rsg = ggn_st[(b*8+rt)*2+1];
    int rb = rt*16 + half8;
    v8h a8;
#pragma unroll
    for (int j = 0; j < 8; ++j) {
      int row = rb + j;
      float v = (acc[j] - mu)*rsg*ggnw[row] + ggnb[row];
      a8[j] = (_Float16)fmaxf(v, 0.f);
    }
    *(v8h*)(&actT[wave][kcol*AST + rb]) = a8;
  }
  __syncthreads();

  // attn = g2 * act + b -> scale, mask, softmax over K, weighted sum with (value + pos)
  bool mok = mask[((size_t)b*MPTS + m)*KNN + kcol] != 0;
  const float scale = 0.088388347648318447f; // 1/sqrt(128)
  const _Float16* vrow = v_h + ((size_t)b*NPTS + id_k)*HID;
#pragma unroll
  for (int s = 0; s < 4; ++s) bf[s] = load_bfrag_T(actT[wave], AST, s*32, lane);
#pragma unroll
  for (int rt = 0; rt < 8; ++rt) {
    v8f acc;
#pragma unroll
    for (int j = 0; j < 8; ++j) acc[j] = g2b[rt*16 + half8 + j];
#pragma unroll
    for (int s = 0; s < 4; ++s) {
      v16h a = load_afrag_h(wh_g2, HID, rt*16, s*32, lane);
      acc = __builtin_amdgcn_wmma_f32_16x16x32_f16(false, a, false, bf[s], (short)0, acc, false, false);
    }
    int rb = rt*16 + half8;
    v8h v8v = *(const v8h*)(vrow + rb);
#pragma unroll
    for (int j = 0; j < 8; ++j) {
      float a = acc[j] * scale;
      a = mok ? a : -3.402823466e38f;
      float mx = wmax16(a);
      float eV = __expf(a - mx);
      float sV = wred16(eV);
      float attn = eV / sV;
      float w = (float)v8v[j] + pacc[rt][j];
      float r = wred16(attn * w);
      if (kcol == 0) res_s[wave][rb + j] = r;
    }
  }
  __syncthreads();

  // post conv (64x128) + residual
  for (int o = tid; o < 8*FDIM; o += 256) {
    int p = o >> 6, ch = o & 63;
    int mp = blockIdx.x*8 + p;
    float accv = postb[ch];
    const float* wr = postw + ch*HID;
    const float* rp = res_s[p];
#pragma unroll 16
    for (int i = 0; i < HID; ++i) accv += wr[i]*rp[i];
    accv += q_feats[((size_t)b*FDIM + ch)*MPTS + mp];
    out[((size_t)b*FDIM + ch)*MPTS + mp] = accv;
  }
}

extern "C" void kernel_launch(void* const* d_in, const int* in_sizes, int n_in,
                              void* d_out, int out_size, void* d_ws, size_t ws_size,
                              hipStream_t stream)
{
  (void)in_sizes; (void)n_in; (void)out_size; (void)ws_size;
  const float* q_xyzs  = (const float*)d_in[0];
  const float* k_xyzs  = (const float*)d_in[1];
  const float* q_feats = (const float*)d_in[2];
  const float* k_feats = (const float*)d_in[3];
  const float* v_feats = (const float*)d_in[4];
  const int*   knn_idx = (const int*)d_in[5];
  const unsigned char* mask = (const unsigned char*)d_in[6];
  const float* wq_w = (const float*)d_in[7];   const float* wq_b = (const float*)d_in[8];
  const float* wk_w = (const float*)d_in[9];   const float* wk_b = (const float*)d_in[10];
  const float* wv_w = (const float*)d_in[11];  const float* wv_b = (const float*)d_in[12];
  const float* d1_w = (const float*)d_in[13];  const float* d1_b = (const float*)d_in[14];
  const float* dgn_w = (const float*)d_in[15]; const float* dgn_b = (const float*)d_in[16];
  const float* d2_w = (const float*)d_in[17];  const float* d2_b = (const float*)d_in[18];
  const float* g1_w = (const float*)d_in[19];  const float* g1_b = (const float*)d_in[20];
  const float* ggn_w = (const float*)d_in[21]; const float* ggn_b = (const float*)d_in[22];
  const float* g2_w = (const float*)d_in[23];  const float* g2_b = (const float*)d_in[24];
  const float* post_w = (const float*)d_in[25];const float* post_b = (const float*)d_in[26];
  float* out = (float*)d_out;

  char* ws = (char*)d_ws;
  size_t off = 0;
  _Float16* wh = (_Float16*)(ws + off);
  const int WSQ = HID*FDIM, WSB = HID*HID;
  const int WTOT = 3*WSQ + 3*WSB; // 73728
  off += (size_t)WTOT * 2;
  _Float16* q_h   = (_Float16*)(ws + off); off += (size_t)BATCH*MPTS*HID*2;
  _Float16* k_h   = (_Float16*)(ws + off); off += (size_t)BATCH*NPTS*HID*2;
  _Float16* v_h   = (_Float16*)(ws + off); off += (size_t)BATCH*NPTS*HID*2;
  size_t stats_off = off;
  float* relstats = (float*)(ws + off); off += (size_t)BATCH*9*4;
  float* dgn_st   = (float*)(ws + off); off += (size_t)BATCH*8*2*4;
  float* ggn_sums = (float*)(ws + off); off += (size_t)BATCH*8*2*4;
  float* ggn_st   = (float*)(ws + off); off += (size_t)BATCH*8*2*4;

  const _Float16* wh_q  = wh;
  const _Float16* wh_k  = wh + WSQ;
  const _Float16* wh_v  = wh + 2*WSQ;
  const _Float16* wh_d2 = wh + 3*WSQ;
  const _Float16* wh_g1 = wh + 3*WSQ + WSB;
  const _Float16* wh_g2 = wh + 3*WSQ + 2*WSB;

  hipMemsetAsync(ws + stats_off, 0, off - stats_off, stream);

  k_cvtw<<<(WTOT + 255)/256, 256, 0, stream>>>(wq_w, wk_w, wv_w, d2_w, g1_w, g2_w, wh);
  k_qkv<<<dim3(NPTS/16, BATCH, 3), 256, 0, stream>>>(
      wh_q, wq_b, q_feats, q_h,
      wh_k, wk_b, k_feats, k_h,
      wh_v, wv_b, v_feats, v_h);
  k_relstats<<<dim3(128, BATCH), 256, 0, stream>>>(q_xyzs, k_xyzs, knn_idx, relstats);
  k_dgn<<<1, 32, 0, stream>>>(relstats, d1_w, d1_b, dgn_st);
  k_stats<<<dim3(MPTS/8, BATCH), 256, 0, stream>>>(
      q_xyzs, k_xyzs, knn_idx, d1_w, d1_b, dgn_w, dgn_b,
      wh_d2, d2_b, wh_g1, g1_b,
      q_h, k_h, dgn_st, ggn_sums);
  k_ggnfin<<<1, 32, 0, stream>>>(ggn_sums, ggn_st);
  k_final<<<dim3(MPTS/8, BATCH), 256, 0, stream>>>(
      q_xyzs, k_xyzs, knn_idx, mask,
      d1_w, d1_b, dgn_w, dgn_b,
      wh_d2, d2_b, wh_g1, g1_b,
      ggn_w, ggn_b, wh_g2, g2_b,
      post_w, post_b, q_feats,
      q_h, k_h, v_h, dgn_st, ggn_st, out);
}
